// EncoderLayer_46153718563354
// MI455X (gfx1250) — matrix operand order repaired and checked by reading, not compiled, not run on hardware
//
#include <hip/hip_runtime.h>
#include <hip/hip_bf16.h>

// ---------------------------------------------------------------------------
// MI455X (gfx1250) transformer encoder layer.
// - All GEMM-shaped math on v_wmma_f32_16x16x32_bf16 (wave32, f32 accumulate).
// - LDS staging via GLOBAL_LOAD_ASYNC_TO_LDS_B128 + s_wait_asynccnt
//   (double-buffered); builtin takes int4 pointers in AS1 (global) / AS3 (LDS).
// - Softmax row reductions on v_permlane16_b32 (VALU) instead of ds_bpermute.
// ---------------------------------------------------------------------------

typedef __bf16 bf16;
typedef __attribute__((ext_vector_type(16))) __bf16 v16bf;
typedef __attribute__((ext_vector_type(8)))  __bf16 bf16x8;
typedef __attribute__((ext_vector_type(8)))  float  v8f;

#define WMMA_BF16(a, b, c) \
  __builtin_amdgcn_wmma_f32_16x16x32_bf16(false, (a), false, (b), (short)0, (c), false, false)

static constexpr int D_  = 768;
static constexpr int H_  = 12;
static constexpr int DK_ = 64;
static constexpr int FF_ = 3072;
static constexpr int B_  = 4;
static constexpr int S_  = 2048;
static constexpr int NT_ = B_ * S_;   // 8192 token rows

// --------------------------- async global->LDS -----------------------------
#if defined(__AMDGCN__) && __has_builtin(__builtin_amdgcn_global_load_async_to_lds_b128)
#define ASYNC_LDS 1
typedef __attribute__((ext_vector_type(4))) int v4i;
typedef __attribute__((address_space(1))) v4i v4i_glb;  // prints as "__device__"
typedef __attribute__((address_space(3))) v4i v4i_lds;  // prints as "__shared__"
#else
#define ASYNC_LDS 0
#endif

__device__ __forceinline__ void copy_b128_to_lds(const bf16* g, bf16* l) {
#if ASYNC_LDS
  __builtin_amdgcn_global_load_async_to_lds_b128((v4i_glb*)g, (v4i_lds*)l, 0, 0);
#else
  *(bf16x8*)l = *(const bf16x8*)g;
#endif
}

__device__ __forceinline__ void async_wait_all() {
#if ASYNC_LDS
#if __has_builtin(__builtin_amdgcn_s_wait_asynccnt)
  __builtin_amdgcn_s_wait_asynccnt(0);
#else
  asm volatile("s_wait_asynccnt 0x0" ::: "memory");
#endif
#endif
}

// --------------------- 16-lane-half xor butterfly --------------------------
template<int MASK>
__device__ __forceinline__ float lane_xor16(float x) {
#if defined(__AMDGCN__) && __has_builtin(__builtin_amdgcn_permlane16)
  constexpr unsigned S0 = MASK == 1 ? 0x67452301u : MASK == 2 ? 0x54761032u :
                          MASK == 4 ? 0x32107654u : 0xFEDCBA98u;
  constexpr unsigned S1 = MASK == 1 ? 0xEFCDAB89u : MASK == 2 ? 0xDCFE98BAu :
                          MASK == 4 ? 0xBA98FEDCu : 0x76543210u;
  return __uint_as_float(__builtin_amdgcn_permlane16(
      __float_as_uint(x), __float_as_uint(x), S0, S1, false, false));
#else
  return __shfl_xor(x, MASK, 32);
#endif
}
__device__ __forceinline__ float red_max16(float x) {
  x = fmaxf(x, lane_xor16<1>(x)); x = fmaxf(x, lane_xor16<2>(x));
  x = fmaxf(x, lane_xor16<4>(x)); x = fmaxf(x, lane_xor16<8>(x));
  return x;
}
__device__ __forceinline__ float red_sum16(float x) {
  x += lane_xor16<1>(x); x += lane_xor16<2>(x);
  x += lane_xor16<4>(x); x += lane_xor16<8>(x);
  return x;
}

// ------------------------------ fragments ----------------------------------
union FragB16 { v16bf v; bf16x8 h[2]; };

// A-fragment (16xK row-major tile in LDS): lane holds row lane%16; element j
// maps to k = j + 8*(lane/16) + (j>=8 ? 8 : 0)  -> two contiguous 16B loads.
__device__ __forceinline__ v16bf frag_a_lds(const bf16* base, int ld) {
  const int lane = threadIdx.x & 31;
  const bf16* p  = base + (lane & 15) * ld + (lane >> 4) * 8;
  FragB16 f;
  f.h[0] = *(const bf16x8*)(p);
  f.h[1] = *(const bf16x8*)(p + 16);
  return f.v;
}

// B-fragment (32x16 K-major tile in LDS): lane holds column lane%16; element j
// maps to k = j + 8*(lane/16) + (j>=8 ? 8 : 0) (strided gather).
__device__ __forceinline__ v16bf frag_b_lds(const bf16* base, int ld) {
  const int lane = threadIdx.x & 31;
  const bf16* p  = base + ((lane >> 4) * 8) * ld + (lane & 15);
  FragB16 f;
  #pragma unroll
  for (int s = 0; s < 16; ++s) f.v[s] = p[(s < 8 ? s : s + 8) * ld];
  return f.v;
}

// ---------------------------------------------------------------------------
// fp32 -> bf16 cast
// ---------------------------------------------------------------------------
__global__ __launch_bounds__(256)
void cast_f32_to_bf16(const float* __restrict__ in, bf16* __restrict__ out, int n) {
  for (int i = blockIdx.x * 256 + threadIdx.x; i < n; i += gridDim.x * 256)
    out[i] = (bf16)in[i];
}

// ---------------------------------------------------------------------------
// Tiled WMMA GEMM: C[MxN] = A[MxK](bf16) @ B[KxN](bf16) + bias (+resid)(+relu)
// BM=64 BN=256 BK=32, 256 threads = 8 waves (2x4), wave tile 32x64 = 2x4 wmma.
// Double-buffered async LDS staging: tile k+1 streams while tile k computes.
// ---------------------------------------------------------------------------
template<bool RELU, bool RESID, bool OUT_BF16>
__global__ __launch_bounds__(256)
void gemm_bf16_wmma(const bf16* __restrict__ A, const bf16* __restrict__ Bm,
                    const float* __restrict__ bias, const float* __restrict__ resid,
                    float* __restrict__ Cf, bf16* __restrict__ Cb,
                    int M, int N, int K)
{
  __shared__ __attribute__((aligned(16))) bf16 sA[2][64 * 32];
  __shared__ __attribute__((aligned(16))) bf16 sB[2][32 * 256];

  const int t    = threadIdx.x;
  const int lane = t & 31;
  const int wv   = t >> 5;
  const int wm   = (wv & 1) * 32;   // wave row offset in block tile
  const int wn   = (wv >> 1) * 64;  // wave col offset in block tile
  const int m0   = blockIdx.y * 64;
  const int n0   = blockIdx.x * 256;

  v8f acc[2][4] = {};

  const int ar = t >> 2, ac = (t & 3) * 8;    // A staging: 8 elems / thread
  const int br = t >> 3, bc = (t & 7) * 32;   // B staging: 32 elems / thread

  auto stage = [&](int buf, int k0) {
    copy_b128_to_lds(&A[(size_t)(m0 + ar) * K + k0 + ac], &sA[buf][ar * 32 + ac]);
    #pragma unroll
    for (int u = 0; u < 4; ++u)
      copy_b128_to_lds(&Bm[(size_t)(k0 + br) * N + n0 + bc + u * 8],
                       &sB[buf][br * 256 + bc + u * 8]);
  };

  const int nk = K >> 5;
  stage(0, 0);
  for (int kt = 0; kt < nk; ++kt) {
    async_wait_all();
    __syncthreads();                       // current buffer resident, prior reads done
    if (kt + 1 < nk) stage((kt + 1) & 1, (kt + 1) << 5);
    if (kt + 2 < nk) {                     // speculative prefetch of tile after next
      __builtin_prefetch(&A[(size_t)(m0 + ar) * K + ((kt + 2) << 5) + ac], 0, 1);
      __builtin_prefetch(&Bm[(size_t)(((kt + 2) << 5) + br) * N + n0 + bc], 0, 1);
    }
    const bf16* cA = sA[kt & 1];
    const bf16* cB = sB[kt & 1];
    v16bf a0 = frag_a_lds(&cA[(wm + 0)  * 32], 32);
    v16bf a1 = frag_a_lds(&cA[(wm + 16) * 32], 32);
    #pragma unroll
    for (int j = 0; j < 4; ++j) {
      v16bf b = frag_b_lds(&cB[wn + j * 16], 256);
      acc[0][j] = WMMA_BF16(a0, b, acc[0][j]);
      acc[1][j] = WMMA_BF16(a1, b, acc[1][j]);
    }
  }

  // Epilogue: C element r lives at row r + 8*(lane/16), col lane%16.
  const int hf = lane >> 4, cn = lane & 15;
  #pragma unroll
  for (int i = 0; i < 2; ++i)
    #pragma unroll
    for (int j = 0; j < 4; ++j)
      #pragma unroll
      for (int r = 0; r < 8; ++r) {
        const int gm = m0 + wm + i * 16 + r + hf * 8;
        const int gn = n0 + wn + j * 16 + cn;
        float v = acc[i][j][r] + bias[gn];
        if (RELU)  v = fmaxf(v, 0.0f);
        if (RESID) v += resid[(size_t)gm * N + gn];
        if (OUT_BF16) Cb[(size_t)gm * N + gn] = (bf16)v;
        else          Cf[(size_t)gm * N + gn] = v;
      }
}

// ---------------------------------------------------------------------------
// Flash attention, one (b,h) x 64 query rows per block; 4 waves x 16 rows.
// Per 32-key step/wave: 4 score WMMAs (Q·K^T, dk=64) + 4 context WMMAs (P·V).
// K/V tiles double-buffered (V via async-to-LDS); softmax stats reduced with
// v_permlane16 across the 16-lane half that holds each row's 16 columns.
// ---------------------------------------------------------------------------
__global__ __launch_bounds__(128)
void flash_attn_wmma(const bf16* __restrict__ Q, const bf16* __restrict__ Kt,
                     const bf16* __restrict__ V, const int* __restrict__ mask,
                     bf16* __restrict__ ctx)
{
  __shared__ __attribute__((aligned(16))) bf16 sQ[64 * 64];         // [qrow][dk]
  __shared__ __attribute__((aligned(16))) bf16 sKT[2][64 * 32];     // [dk][kpos]
  __shared__ __attribute__((aligned(16))) bf16 sV[2][32 * 64];      // [kpos][dk]
  __shared__ __attribute__((aligned(16))) bf16 sP[4][16 * 32];      // per-wave P

  const int t    = threadIdx.x;
  const int lane = t & 31;
  const int wv   = t >> 5;
  const int b    = blockIdx.y / H_;
  const int h    = blockIdx.y % H_;
  const int q0   = blockIdx.x * 64;
  const size_t rowbase = (size_t)b * S_;
  const int c0   = h * DK_;

  const int hf  = lane >> 4, cn = lane & 15;
  const int kp  = t >> 2, dk0 = (t & 3) * 16;

  auto stage_kv = [&](int buf, int kb) {
    // V chunk row-major (async path)
    const bf16* vsrc = &V[(rowbase + kb + kp) * D_ + c0 + dk0];
    copy_b128_to_lds(vsrc,     &sV[buf][kp * 64 + dk0]);
    copy_b128_to_lds(vsrc + 8, &sV[buf][kp * 64 + dk0 + 8]);
    // K chunk transposed to [dk][kpos] (per-element, stays on the DS pipe)
    union { bf16x8 v[2]; bf16 e[16]; } tmp;
    const bf16* ksrc = &Kt[(rowbase + kb + kp) * D_ + c0 + dk0];
    tmp.v[0] = *(const bf16x8*)(ksrc);
    tmp.v[1] = *(const bf16x8*)(ksrc + 8);
    #pragma unroll
    for (int u = 0; u < 16; ++u) sKT[buf][(dk0 + u) * 32 + kp] = tmp.e[u];
  };

  { // stage Q tile (64x64) into LDS (async path)
    const int r = t >> 1, cc = (t & 1) * 32;
    const bf16* src = &Q[(rowbase + q0 + r) * D_ + c0 + cc];
    bf16* dst = &sQ[r * 64 + cc];
    #pragma unroll
    for (int u = 0; u < 4; ++u) copy_b128_to_lds(src + u * 8, dst + u * 8);
  }
  stage_kv(0, 0);
  async_wait_all();
  __syncthreads();

  v16bf qf[2];
  qf[0] = frag_a_lds(&sQ[(wv * 16) * 64 + 0],  64);
  qf[1] = frag_a_lds(&sQ[(wv * 16) * 64 + 32], 64);

  float mrow[8], lrow[8];
  v8f o[4] = {};
  #pragma unroll
  for (int r = 0; r < 8; ++r) { mrow[r] = -3.0e38f; lrow[r] = 0.0f; }

  for (int kb = 0; kb < S_; kb += 32) {
    const int buf = (kb >> 5) & 1;
    if (kb + 32 < S_) stage_kv(buf ^ 1, kb + 32);  // stream next tile during compute

    v8f s0 = {}, s1 = {};
    #pragma unroll
    for (int kc = 0; kc < 2; ++kc) {  // dk chunks of 32
      v16bf bk0 = frag_b_lds(&sKT[buf][kc * 32 * 32 + 0],  32);
      v16bf bk1 = frag_b_lds(&sKT[buf][kc * 32 * 32 + 16], 32);
      s0 = WMMA_BF16(qf[kc], bk0, s0);
      s1 = WMMA_BF16(qf[kc], bk1, s1);
    }

    const float scale = 0.125f;  // 1/sqrt(DK)
    const int mk0 = mask[(size_t)b * S_ + kb + cn];
    const int mk1 = mask[(size_t)b * S_ + kb + 16 + cn];

    #pragma unroll
    for (int r = 0; r < 8; ++r) {
      const float v0 = mk0 ? s0[r] * scale : -1.0e9f;
      const float v1 = mk1 ? s1[r] * scale : -1.0e9f;
      const float mx = red_max16(fmaxf(v0, v1));
      const float mnew = fmaxf(mrow[r], mx);
      const float corr = __expf(mrow[r] - mnew);
      const float p0 = __expf(v0 - mnew);
      const float p1 = __expf(v1 - mnew);
      lrow[r] = lrow[r] * corr + red_sum16(p0 + p1);
      mrow[r] = mnew;
      #pragma unroll
      for (int j = 0; j < 4; ++j) o[j][r] *= corr;
      // C-layout -> row-major P tile in LDS (re-shaped to A-layout on re-read)
      sP[wv][(r + hf * 8) * 32 + cn]      = (bf16)p0;
      sP[wv][(r + hf * 8) * 32 + 16 + cn] = (bf16)p1;
    }
    __syncthreads();   // P visible for A-fragment re-read

    v16bf ap = frag_a_lds(&sP[wv][0], 32);
    #pragma unroll
    for (int j = 0; j < 4; ++j) {   // O(16x64) += P(16x32) @ V(32x64)
      v16bf bv = frag_b_lds(&sV[buf][j * 16], 64);
      o[j] = WMMA_BF16(ap, bv, o[j]);
    }

    async_wait_all();  // next K/V tile resident; all waves done with this one
    __syncthreads();
  }

  #pragma unroll
  for (int r = 0; r < 8; ++r) {
    const float inv = 1.0f / lrow[r];
    const size_t grow = (rowbase + q0 + wv * 16 + r + hf * 8) * (size_t)D_;
    #pragma unroll
    for (int j = 0; j < 4; ++j)
      ctx[grow + c0 + j * 16 + cn] = (bf16)(o[j][r] * inv);
  }
}

// ---------------------------------------------------------------------------
// LayerNorm over D=768; emits fp32 (residual path) and optional bf16 copy.
// ---------------------------------------------------------------------------
__global__ __launch_bounds__(256)
void layernorm_kernel(const float* __restrict__ in, const float* __restrict__ g,
                      const float* __restrict__ bta, float* __restrict__ outf,
                      bf16* __restrict__ outb)
{
  __shared__ float red[256];
  const int row = blockIdx.x, tid = threadIdx.x;
  const float* x = in + (size_t)row * D_;

  float s = 0.0f;
  for (int i = tid; i < D_; i += 256) s += x[i];
  red[tid] = s; __syncthreads();
  for (int st = 128; st > 0; st >>= 1) { if (tid < st) red[tid] += red[tid + st]; __syncthreads(); }
  const float mu = red[0] * (1.0f / D_);
  __syncthreads();

  float v = 0.0f;
  for (int i = tid; i < D_; i += 256) { const float d = x[i] - mu; v += d * d; }
  red[tid] = v; __syncthreads();
  for (int st = 128; st > 0; st >>= 1) { if (tid < st) red[tid] += red[tid + st]; __syncthreads(); }
  const float rstd = rsqrtf(red[0] * (1.0f / D_) + 1e-5f);
  __syncthreads();

  for (int i = tid; i < D_; i += 256) {
    const float y = (x[i] - mu) * rstd * g[i] + bta[i];
    outf[(size_t)row * D_ + i] = y;
    if (outb) outb[(size_t)row * D_ + i] = (bf16)y;
  }
}

// ---------------------------------------------------------------------------
// Launch
// ---------------------------------------------------------------------------
extern "C" void kernel_launch(void* const* d_in, const int* in_sizes, int n_in,
                              void* d_out, int out_size, void* d_ws, size_t ws_size,
                              hipStream_t stream) {
  (void)in_sizes; (void)n_in; (void)out_size; (void)ws_size;

  const float* emb  = (const float*)d_in[0];
  const int*   mask = (const int*)  d_in[1];
  const float* Wq   = (const float*)d_in[2];   const float* bq = (const float*)d_in[3];
  const float* Wk   = (const float*)d_in[4];   const float* bk = (const float*)d_in[5];
  const float* Wv   = (const float*)d_in[6];   const float* bv = (const float*)d_in[7];
  const float* Wo   = (const float*)d_in[8];   const float* bo = (const float*)d_in[9];
  const float* W1   = (const float*)d_in[10];  const float* b1 = (const float*)d_in[11];
  const float* W2   = (const float*)d_in[12];  const float* b2 = (const float*)d_in[13];
  const float* ln_g = (const float*)d_in[14];  const float* ln_b = (const float*)d_in[15];
  float* out = (float*)d_out;

  char* ws = (char*)d_ws;
  size_t off = 0;
  auto alloc = [&](size_t bytes) { size_t o = off; off = (off + bytes + 255) & ~(size_t)255; return o; };

  bf16* Xb   = (bf16*)(ws + alloc((size_t)NT_ * D_  * 2));
  bf16* Wqb  = (bf16*)(ws + alloc((size_t)D_  * D_  * 2));
  bf16* Wkb  = (bf16*)(ws + alloc((size_t)D_  * D_  * 2));
  bf16* Wvb  = (bf16*)(ws + alloc((size_t)D_  * D_  * 2));
  bf16* Wob  = (bf16*)(ws + alloc((size_t)D_  * D_  * 2));
  bf16* W1b  = (bf16*)(ws + alloc((size_t)D_  * FF_ * 2));
  bf16* W2b  = (bf16*)(ws + alloc((size_t)FF_ * D_  * 2));
  bf16* Qb   = (bf16*)(ws + alloc((size_t)NT_ * D_  * 2));
  bf16* Kb   = (bf16*)(ws + alloc((size_t)NT_ * D_  * 2));
  bf16* Vb   = (bf16*)(ws + alloc((size_t)NT_ * D_  * 2));
  bf16* Ctxb = (bf16*)(ws + alloc((size_t)NT_ * D_  * 2));
  float* res = (float*)(ws + alloc((size_t)NT_ * D_ * 4));  // reused for FFN residual sum
  float* xf  = (float*)(ws + alloc((size_t)NT_ * D_ * 4));
  bf16* xb   = (bf16*)(ws + alloc((size_t)NT_ * D_  * 2));
  bf16* hb   = (bf16*)(ws + alloc((size_t)NT_ * FF_ * 2));

  auto cast = [&](const float* src, bf16* dst, int n) {
    cast_f32_to_bf16<<<(n + 255) / 256, 256, 0, stream>>>(src, dst, n);
  };
  cast(emb, Xb, NT_ * D_);
  cast(Wq, Wqb, D_ * D_);  cast(Wk, Wkb, D_ * D_);
  cast(Wv, Wvb, D_ * D_);  cast(Wo, Wob, D_ * D_);
  cast(W1, W1b, D_ * FF_); cast(W2, W2b, FF_ * D_);

  // QKV projections -> bf16 [NT, D]
  {
    dim3 g(D_ / 256, NT_ / 64);
    gemm_bf16_wmma<false, false, true><<<g, 256, 0, stream>>>(Xb, Wqb, bq, nullptr, nullptr, Qb, NT_, D_, D_);
    gemm_bf16_wmma<false, false, true><<<g, 256, 0, stream>>>(Xb, Wkb, bk, nullptr, nullptr, Kb, NT_, D_, D_);
    gemm_bf16_wmma<false, false, true><<<g, 256, 0, stream>>>(Xb, Wvb, bv, nullptr, nullptr, Vb, NT_, D_, D_);
  }

  // Attention -> bf16 context [NT, D]
  flash_attn_wmma<<<dim3(S_ / 64, B_ * H_), 128, 0, stream>>>(Qb, Kb, Vb, mask, Ctxb);

  // Output projection + residual(embedding) -> fp32 res
  gemm_bf16_wmma<false, true, false><<<dim3(D_ / 256, NT_ / 64), 256, 0, stream>>>(
      Ctxb, Wob, bo, emb, res, nullptr, NT_, D_, D_);

  // LN1 -> x (fp32 residual copy + bf16 GEMM input)
  layernorm_kernel<<<NT_, 256, 0, stream>>>(res, ln_g, ln_b, xf, xb);

  // FFN1: relu(x @ W1 + b1) -> bf16 h
  gemm_bf16_wmma<true, false, true><<<dim3(FF_ / 256, NT_ / 64), 256, 0, stream>>>(
      xb, W1b, b1, nullptr, nullptr, hb, NT_, FF_, D_);

  // FFN2: h @ W2 + b2 + x -> fp32 res (reuse)
  gemm_bf16_wmma<false, true, false><<<dim3(D_ / 256, NT_ / 64), 256, 0, stream>>>(
      hb, W2b, b2, xf, res, nullptr, NT_, D_, FF_);

  // LN2 -> final output (fp32)
  layernorm_kernel<<<NT_, 256, 0, stream>>>(res, ln_g, ln_b, out, nullptr);
}
